// SequenceExtrapolator_81570018885988
// MI455X (gfx1250) — matrix operand order repaired and checked by reading, not compiled, not run on hardware
//
#include <hip/hip_runtime.h>

typedef __attribute__((ext_vector_type(16))) __bf16 bf16x16;
typedef __attribute__((ext_vector_type(8)))  __bf16 bf16x8;
typedef __attribute__((ext_vector_type(8)))  float  f32x8;

static constexpr int Bq  = 4;
static constexpr int Sq  = 2048;
static constexpr int Dq  = 1024;
static constexpr int Hq  = 16;
static constexpr int HDq = 64;
static constexpr int Lq  = 4;
static constexpr int FFq = 4096;
static constexpr int WIN = 256;

// ---------------------------------------------------------------------------
// helpers
// ---------------------------------------------------------------------------
static __device__ __forceinline__ bf16x16 cat8(bf16x8 lo, bf16x8 hi) {
  return __builtin_shufflevector(lo, hi, 0,1,2,3,4,5,6,7,8,9,10,11,12,13,14,15);
}
// A-fragment: 8 bf16 at p, 8 bf16 at p+16 (p pre-offset by kbA)
static __device__ __forceinline__ bf16x16 ld_afrag(const __bf16* p) {
  return cat8(*(const bf16x8*)p, *(const bf16x8*)(p + 16));
}
// B-fragment: 8 bf16 at p, 8 bf16 at p+16 (p pre-offset by kbB)
static __device__ __forceinline__ bf16x16 ld_bfrag(const __bf16* p) {
  return cat8(*(const bf16x8*)p, *(const bf16x8*)(p + 16));
}
static __device__ __forceinline__ f32x8 wmma_bf16(bf16x16 a, bf16x16 b, f32x8 c) {
  return __builtin_amdgcn_wmma_f32_16x16x32_bf16(false, a, false, b, (short)0, c,
                                                 false, false);
}
// VALU rotate-reduce max over 16-lane row groups via DPP row_ror (no LDS)
template <int CTRL>
static __device__ __forceinline__ float fmax_ror(float x) {
  const int v = __builtin_amdgcn_update_dpp(0, __builtin_bit_cast(int, x),
                                            CTRL, 0xF, 0xF, true);
  return fmaxf(x, __builtin_bit_cast(float, v));
}
static __device__ __forceinline__ float rowmax16(float x) {
  x = fmax_ror<0x121>(x);   // row_ror:1
  x = fmax_ror<0x122>(x);   // row_ror:2
  x = fmax_ror<0x124>(x);   // row_ror:4
  x = fmax_ror<0x128>(x);   // row_ror:8
  return x;
}

// ---------------------------------------------------------------------------
// RMSNorm (fp32 in) -> bf16 out, one 256-thread block per row of D=1024
// ---------------------------------------------------------------------------
__global__ __launch_bounds__(256)
void rmsnorm_to_bf16(const float* __restrict__ x, __bf16* __restrict__ h) {
  const long row = blockIdx.x;
  const float* xr = x + row * Dq;
  float ss = 0.f;
#pragma unroll
  for (int i = 0; i < Dq / 256; ++i) {
    const float v = xr[threadIdx.x + i * 256];
    ss += v * v;
  }
#pragma unroll
  for (int off = 16; off > 0; off >>= 1) ss += __shfl_xor(ss, off, 32);
  __shared__ float wsum[8];
  if ((threadIdx.x & 31) == 0) wsum[threadIdx.x >> 5] = ss;
  __syncthreads();
  float tot = 0.f;
#pragma unroll
  for (int w = 0; w < 8; ++w) tot += wsum[w];
  const float rinv = rsqrtf(tot * (1.0f / Dq) + 1e-6f);
  __bf16* hr = h + row * Dq;
#pragma unroll
  for (int i = 0; i < Dq / 256; ++i) {
    const int idx = threadIdx.x + i * 256;
    hr[idx] = (__bf16)(xr[idx] * rinv);
  }
}

// ---------------------------------------------------------------------------
// Weight prep: W[K][N] fp32 -> Wt[N][K] bf16 (LDS-tiled transpose)
// ---------------------------------------------------------------------------
__global__ __launch_bounds__(256)
void transpose_to_bf16(const float* __restrict__ W, __bf16* __restrict__ Wt,
                       int K, int N) {
  __shared__ float tile[64][65];
  const int k0 = blockIdx.x * 64;
  const int n0 = blockIdx.y * 64;
  const int tx = threadIdx.x & 63;
  const int ty = threadIdx.x >> 6;
#pragma unroll
  for (int i = 0; i < 16; ++i) {
    const int k = ty + i * 4;
    tile[k][tx] = W[(long)(k0 + k) * N + n0 + tx];
  }
  __syncthreads();
#pragma unroll
  for (int i = 0; i < 16; ++i) {
    const int n = ty + i * 4;
    Wt[(long)(n0 + n) * K + k0 + tx] = (__bf16)tile[tx][n];
  }
}

__global__ __launch_bounds__(256)
void convert_bf16(const float* __restrict__ in, __bf16* __restrict__ out) {
  const long i = ((long)blockIdx.x * 256 + threadIdx.x) * 4;
#pragma unroll
  for (int j = 0; j < 4; ++j) out[i + j] = (__bf16)in[i + j];
}

// ---------------------------------------------------------------------------
// RoPE + repack [B,S,D] fp32 -> [B,H,S,HD] bf16, with output scale (Q: 1/8)
// ---------------------------------------------------------------------------
__global__ __launch_bounds__(256)
void rope_pack(const float* __restrict__ t, const float* __restrict__ cosb,
               const float* __restrict__ sinb, __bf16* __restrict__ outp,
               float oscale) {
  const long row = blockIdx.x;               // b*S + s
  const int b = (int)(row / Sq);
  const int s = (int)(row % Sq);
  const float* tr = t + row * Dq;
#pragma unroll
  for (int it = 0; it < Dq / 256; ++it) {
    const int d = threadIdx.x + it * 256;
    const int hh = d >> 6, j = d & 63;
    const float c  = cosb[s * HDq + j];
    const float sn = sinb[s * HDq + j];
    const float rot = (j < 32) ? -tr[hh * 64 + j + 32] : tr[hh * 64 + j - 32];
    const float v = (tr[d] * c + rot * sn) * oscale;
    outp[(((long)b * Hq + hh) * Sq + s) * HDq + j] = (__bf16)v;
  }
}

// ---------------------------------------------------------------------------
// Pack V: [B,S,D] fp32 -> transposed [B,H,HD,S] bf16 (LDS-tiled)
// ---------------------------------------------------------------------------
__global__ __launch_bounds__(256)
void pack_vT(const float* __restrict__ t, __bf16* __restrict__ vt) {
  __shared__ float tile[64][65];
  const int s0 = blockIdx.x * 64;
  const int h  = blockIdx.y;
  const int b  = blockIdx.z;
  const int tx = threadIdx.x & 63;
  const int ty = threadIdx.x >> 6;
#pragma unroll
  for (int i = 0; i < 16; ++i) {
    const int s = ty + i * 4;
    tile[s][tx] = t[((long)b * Sq + s0 + s) * Dq + h * HDq + tx];
  }
  __syncthreads();
#pragma unroll
  for (int i = 0; i < 16; ++i) {
    const int j = ty + i * 4;   // hd
    vt[(((long)b * Hq + h) * HDq + j) * Sq + s0 + tx] = (__bf16)tile[tx][j];
  }
}

// ---------------------------------------------------------------------------
// LDS-free bf16 WMMA GEMM: C[M,N] = A_bf16[M,K] x Bt_bf16[N,K]^T
// 256 threads / 8 waves; block tile 64x128; each wave 32x32 (4 WMMAs).
// Both operands stream straight from L2 as b128 fragment loads.
// EPI: 0 = fp32 store, 1 = +residual fp32, 2 = SiLU->bf16, 3 = tanh fp32
// ---------------------------------------------------------------------------
template <int EPI>
__global__ __launch_bounds__(256)
void gemm_wmma(const __bf16* __restrict__ A, const __bf16* __restrict__ Bt,
               int M, int N, int K,
               float* __restrict__ Cf, __bf16* __restrict__ Cb,
               const float* __restrict__ res) {
  const int lane = threadIdx.x & 31;
  const int wave = threadIdx.x >> 5;
  const int mrow = lane & 15;
  const int hi  = (lane < 16) ? 0 : 8;
  const int kbA = (lane < 16) ? 0 : 8;
  const int kbB = (lane < 16) ? 0 : 8;
  const long cm = (long)blockIdx.x * 64  + (wave & 1) * 32;
  const long cn = (long)blockIdx.y * 128 + (wave >> 1) * 32;

  const __bf16* a0 = A  + (cm + mrow)      * (long)K + kbA;
  const __bf16* a1 = A  + (cm + 16 + mrow) * (long)K + kbA;
  const __bf16* b0 = Bt + (cn + mrow)      * (long)K + kbB;
  const __bf16* b1 = Bt + (cn + 16 + mrow) * (long)K + kbB;

  f32x8 acc00 = {}, acc01 = {}, acc10 = {}, acc11 = {};
  for (int kk = 0; kk < K; kk += 32) {
    __builtin_prefetch(a0 + kk + 64, 0, 3);
    __builtin_prefetch(a1 + kk + 64, 0, 3);
    __builtin_prefetch(b0 + kk + 64, 0, 3);
    __builtin_prefetch(b1 + kk + 64, 0, 3);
    const bf16x16 af0 = ld_afrag(a0 + kk);
    const bf16x16 af1 = ld_afrag(a1 + kk);
    const bf16x16 bv0 = ld_bfrag(b0 + kk);
    const bf16x16 bv1 = ld_bfrag(b1 + kk);
    acc00 = wmma_bf16(af0, bv0, acc00);
    acc01 = wmma_bf16(af0, bv1, acc01);
    acc10 = wmma_bf16(af1, bv0, acc10);
    acc11 = wmma_bf16(af1, bv1, acc11);
  }

  auto emit = [&](const f32x8& acc, int mo, int no) {
#pragma unroll
    for (int r = 0; r < 8; ++r) {
      const long row = cm + mo + r + hi;
      const long col = cn + no + mrow;
      const long i = row * N + col;
      const float v = acc[r];
      if (EPI == 0)      Cf[i] = v;
      else if (EPI == 1) Cf[i] = v + res[i];
      else if (EPI == 2) Cb[i] = (__bf16)(v / (1.f + __expf(-v)));
      else               Cf[i] = tanhf(v);   // CAP = 1.0
    }
  };
  emit(acc00, 0, 0); emit(acc01, 0, 16); emit(acc10, 16, 0); emit(acc11, 16, 16);
}

// ---------------------------------------------------------------------------
// Sliding-window flash attention.
// Q/K bf16 [B,H,S,HD] (Q pre-scaled by 1/8); V bf16 transposed [B,H,HD,S].
// 128 threads = 4 waves, wave w owns queries [qb0+16w, qb0+16w+16).
// Row-sum via extra ones-column WMMA; row-max via DPP rotate-reduce.
// ---------------------------------------------------------------------------
__global__ __launch_bounds__(128)
void attn_swa(const __bf16* __restrict__ qm, const __bf16* __restrict__ km,
              const __bf16* __restrict__ vt, __bf16* __restrict__ om) {
  __shared__ __align__(16) __bf16 ps[4][16][32];   // per-wave P tile
  const int lane = threadIdx.x & 31;
  const int wave = threadIdx.x >> 5;
  const int h = blockIdx.y, b = blockIdx.z;
  const int qb0 = blockIdx.x * 64;
  const int q0  = qb0 + wave * 16;
  const long bh = ((long)b * Hq + h) * Sq;
  const __bf16* Qp = qm + bh * HDq;
  const __bf16* Kp = km + bh * HDq;
  const __bf16* Vt = vt + bh * HDq;    // [HD][S] for this (b,h)
  const int mrow = lane & 15;
  const int hi  = (lane < 16) ? 0 : 8;
  const int kbA = (lane < 16) ? 0 : 8;
  const int kbB = (lane < 16) ? 0 : 8;

  bf16x16 bones;
#pragma unroll
  for (int j = 0; j < 16; ++j) bones[j] = (__bf16)1.0f;

  // Q A-fragments for hd chunks [0,32) and [32,64)
  const __bf16* qr = Qp + (long)(q0 + mrow) * HDq;
  const bf16x16 aq0 = ld_afrag(qr + kbA);
  const bf16x16 aq1 = ld_afrag(qr + 32 + kbA);

  float mx[8];
  f32x8 oacc[4] = {};
  f32x8 osum = {};
#pragma unroll
  for (int r = 0; r < 8; ++r) mx[r] = -3.0e38f;

  const int ks = (qb0 >= WIN) ? ((qb0 - (WIN - 1)) & ~31) : 0;
  const int ke = qb0 + 63;
  for (int k0 = ks; k0 <= ke; k0 += 32) {
    // ---- scores: two 16-key halves; HD=64 in two K=32 WMMA steps ----
    f32x8 sc0 = {}, sc1 = {};
    {
      const __bf16* kr = Kp + (long)(k0 + mrow) * HDq;
      sc0 = wmma_bf16(aq0, ld_bfrag(kr + kbB), sc0);
      sc0 = wmma_bf16(aq1, ld_bfrag(kr + 32 + kbB), sc0);
    }
    {
      const __bf16* kr = Kp + (long)(k0 + 16 + mrow) * HDq;
      sc1 = wmma_bf16(aq0, ld_bfrag(kr + kbB), sc1);
      sc1 = wmma_bf16(aq1, ld_bfrag(kr + 32 + kbB), sc1);
    }
    // ---- mask + online softmax (DPP row-max, WMMA row-sum) ----
#pragma unroll
    for (int r = 0; r < 8; ++r) {
      const int q = q0 + r + hi;
      const int key0 = k0 + mrow, key1 = k0 + 16 + mrow;
      const bool v0 = (key0 <= q) && (q - key0 < WIN);
      const bool v1 = (key1 <= q) && (q - key1 < WIN);
      const float s0 = v0 ? sc0[r] : -1.0e30f;
      const float s1 = v1 ? sc1[r] : -1.0e30f;
      const float mnew = fmaxf(mx[r], rowmax16(fmaxf(s0, s1)));
      const float sf = __expf(mx[r] - mnew);
      mx[r] = mnew;
      const float e0 = v0 ? __expf(s0 - mnew) : 0.f;
      const float e1 = v1 ? __expf(s1 - mnew) : 0.f;
      osum[r] *= sf;
#pragma unroll
      for (int nt = 0; nt < 4; ++nt) oacc[nt][r] *= sf;
      ps[wave][r + hi][mrow]      = (__bf16)e0;
      ps[wave][r + hi][16 + mrow] = (__bf16)e1;
    }
    __syncthreads();
    // ---- P (LDS transpose) as A-fragment; O += P x V ; l += P x 1 ----
    const bf16x16 ap = ld_afrag(&ps[wave][mrow][0] + kbA);
#pragma unroll
    for (int nt = 0; nt < 4; ++nt) {
      const __bf16* vr = Vt + (long)(nt * 16 + mrow) * Sq + k0 + kbB;
      oacc[nt] = wmma_bf16(ap, ld_bfrag(vr), oacc[nt]);
    }
    osum = wmma_bf16(ap, bones, osum);
    __syncthreads();
  }

#pragma unroll
  for (int r = 0; r < 8; ++r) {
    const int q = q0 + r + hi;
    const float inv = 1.0f / osum[r];
    const long base = ((long)b * Sq + q) * Dq + h * HDq + mrow;
#pragma unroll
    for (int nt = 0; nt < 4; ++nt)
      om[base + nt * 16] = (__bf16)(oacc[nt][r] * inv);
  }
}

// ---------------------------------------------------------------------------
extern "C" void kernel_launch(void* const* d_in, const int* in_sizes, int n_in,
                              void* d_out, int out_size, void* d_ws, size_t ws_size,
                              hipStream_t stream) {
  const float* latents = (const float*)d_in[0];
  const float* cosb    = (const float*)d_in[1];
  const float* sinb    = (const float*)d_in[2];
  const float* wq      = (const float*)d_in[3];
  const float* wk      = (const float*)d_in[4];
  const float* wv      = (const float*)d_in[5];
  const float* wo      = (const float*)d_in[6];
  const float* w1      = (const float*)d_in[7];
  const float* w2      = (const float*)d_in[8];
  const float* dw      = (const float*)d_in[9];
  float* out = (float*)d_out;
  (void)in_sizes; (void)n_in; (void)out_size; (void)ws_size;

  const long M = (long)Bq * Sq;   // 8192
  char* p = (char*)d_ws;
  auto take = [&](size_t bytes) { char* r = p; p += (bytes + 255) & ~(size_t)255; return r; };
  float*  x   = (float*) take((size_t)M * Dq * 4);
  __bf16* hbf = (__bf16*)take((size_t)M * Dq * 2);
  float*  tmp = (float*) take((size_t)M * Dq * 4);
  __bf16* qbf = (__bf16*)take((size_t)M * Dq * 2);
  __bf16* kbf = (__bf16*)take((size_t)M * Dq * 2);
  __bf16* vtb = (__bf16*)take((size_t)M * Dq * 2);
  __bf16* obf = (__bf16*)take((size_t)M * Dq * 2);
  __bf16* h1  = (__bf16*)take((size_t)M * FFq * 2);
  __bf16* wt  = (__bf16*)take((size_t)Lq * 12 * 1024 * 1024 * 2);
  __bf16* dwt = (__bf16*)take((size_t)Dq * Dq * 2);

  const size_t MB1 = 1024 * 1024;

  hipMemcpyAsync(x, latents, (size_t)M * Dq * 4, hipMemcpyDeviceToDevice, stream);

  // ---- weight prep: fp32 -> transposed bf16 (one pass, L2 resident) ----
  const dim3 blk(256);
  for (int l = 0; l < Lq; ++l) {
    __bf16* base = wt + (size_t)l * 12 * MB1;
    transpose_to_bf16<<<dim3(Dq/64, Dq/64), blk, 0, stream>>>(wq + (size_t)l*Dq*Dq, base + 0*MB1, Dq, Dq);
    transpose_to_bf16<<<dim3(Dq/64, Dq/64), blk, 0, stream>>>(wk + (size_t)l*Dq*Dq, base + 1*MB1, Dq, Dq);
    transpose_to_bf16<<<dim3(Dq/64, Dq/64), blk, 0, stream>>>(wv + (size_t)l*Dq*Dq, base + 2*MB1, Dq, Dq);
    transpose_to_bf16<<<dim3(Dq/64, Dq/64), blk, 0, stream>>>(wo + (size_t)l*Dq*Dq, base + 3*MB1, Dq, Dq);
    transpose_to_bf16<<<dim3(Dq/64, FFq/64), blk, 0, stream>>>(w1 + (size_t)l*Dq*FFq, base + 4*MB1, Dq, FFq);
    transpose_to_bf16<<<dim3(FFq/64, Dq/64), blk, 0, stream>>>(w2 + (size_t)l*FFq*Dq, base + 8*MB1, FFq, Dq);
  }
  convert_bf16<<<1024, blk, 0, stream>>>(dw, dwt);   // delta_w^T needs no transpose

  const dim3 gDD((unsigned)(M / 64), Dq / 128);
  const dim3 gDF((unsigned)(M / 64), FFq / 128);
  const dim3 gA(Sq / 64, Hq, Bq);
  const dim3 gR((unsigned)M);

  for (int l = 0; l < Lq; ++l) {
    __bf16* base = wt + (size_t)l * 12 * MB1;
    rmsnorm_to_bf16<<<gR, blk, 0, stream>>>(x, hbf);
    gemm_wmma<0><<<gDD, blk, 0, stream>>>(hbf, base + 0*MB1, (int)M, Dq, Dq, tmp, nullptr, nullptr);
    rope_pack<<<gR, blk, 0, stream>>>(tmp, cosb, sinb, qbf, 0.125f);   // fold 1/sqrt(HD)
    gemm_wmma<0><<<gDD, blk, 0, stream>>>(hbf, base + 1*MB1, (int)M, Dq, Dq, tmp, nullptr, nullptr);
    rope_pack<<<gR, blk, 0, stream>>>(tmp, cosb, sinb, kbf, 1.0f);
    gemm_wmma<0><<<gDD, blk, 0, stream>>>(hbf, base + 2*MB1, (int)M, Dq, Dq, tmp, nullptr, nullptr);
    pack_vT<<<gA, blk, 0, stream>>>(tmp, vtb);
    attn_swa<<<gA, dim3(128), 0, stream>>>(qbf, kbf, vtb, obf);
    gemm_wmma<1><<<gDD, blk, 0, stream>>>(obf, base + 3*MB1, (int)M, Dq, Dq, x, nullptr, x);
    rmsnorm_to_bf16<<<gR, blk, 0, stream>>>(x, hbf);
    gemm_wmma<2><<<gDF, blk, 0, stream>>>(hbf, base + 4*MB1, (int)M, FFq, Dq, nullptr, h1, nullptr);
    gemm_wmma<1><<<gDD, blk, 0, stream>>>(h1, base + 8*MB1, (int)M, Dq, FFq, x, nullptr, x);
  }
  rmsnorm_to_bf16<<<gR, blk, 0, stream>>>(x, hbf);
  gemm_wmma<3><<<gDD, blk, 0, stream>>>(hbf, dwt, (int)M, Dq, Dq, out, nullptr, nullptr);
}
